// MSARowAttentionWithPairBias_63763084476746
// MI455X (gfx1250) — compile-verified, matrix operand order checked by reading, not run on hardware
//
#include <hip/hip_runtime.h>
#include <hip/hip_bf16.h>

typedef _Float16 half_t;
typedef __attribute__((ext_vector_type(16))) _Float16 v16h;
typedef __attribute__((ext_vector_type(8)))  float    v8f;
typedef __attribute__((ext_vector_type(4)))  unsigned int u32x4;
typedef __attribute__((ext_vector_type(8)))  int      i32x8;
typedef __attribute__((ext_vector_type(4)))  int      i32x4;

#define S_SEQ 128
#define R_RES 256
#define CM    256
#define CZ    128
#define CH    32
#define NH    8

#if defined(__has_builtin)
#if __has_builtin(__builtin_amdgcn_tensor_load_to_lds)
#define HAVE_TDM 1
#endif
#endif

// ---------------------------------------------------------------------------
// WMMA 16x16x32 f16 fragment loader (A-layout; B uses identical layout with
// lane = N, since B is consumed as N x K).
// lane l: row = row0 + (l&15), half = l>>4
// element e: k = k0 + 16*(e/8) + 8*half + (e%8)
// -> two contiguous 16-byte chunks per lane (global_load_b128 / ds_load_b128).
// ---------------------------------------------------------------------------
__device__ __forceinline__ v16h load_frag16(const half_t* base, int row0, int ldk, int k0) {
  const int lane = threadIdx.x & 31;
  const int r    = row0 + (lane & 15);
  const int hh   = lane >> 4;
  const half_t* p = base + (size_t)r * ldk + k0 + hh * 8;
  union { v16h v; uint4 q[2]; } u;
  u.q[0] = *(const uint4*)(p);
  u.q[1] = *(const uint4*)(p + 16);
  return u.v;
}

__device__ __forceinline__ v8f wmma_f16(v16h a, v16h b, v8f c) {
  return __builtin_amdgcn_wmma_f32_16x16x32_f16(false, a, false, b, (short)0, c, false, false);
}

__device__ __forceinline__ void wait_tensorcnt0() {
#if defined(__has_builtin)
#if __has_builtin(__builtin_amdgcn_s_wait_tensorcnt)
  __builtin_amdgcn_s_wait_tensorcnt(0);
#else
  asm volatile("s_wait_tensorcnt 0x0" ::: "memory");
#endif
#endif
}

#if defined(HAVE_TDM)
// One 2D TDM transfer: tile == tensor region of dim0 (contiguous, f16 elems)
// by dim1 rows with row stride stride0 (elems), global -> LDS.
// Descriptor per ISA 08 §8.3/8.4; groups 2/3 zero (up-to-2D form).
__device__ __forceinline__ void tdm_stage(const half_t* gsrc, half_t* ldst,
                                          unsigned dim0, unsigned dim1,
                                          unsigned stride0) {
  unsigned long long ga = (unsigned long long)(const void*)gsrc;
  unsigned lds = (unsigned)(unsigned long long)(const void*)ldst; // LDS aperture: low 32b = LDS byte addr
  u32x4 g0;
  g0[0] = 1u;                                                  // count=1, user D#
  g0[1] = lds;                                                 // lds_addr [63:32]
  g0[2] = (unsigned)(ga & 0xffffffffull);                      // global_addr lo
  g0[3] = (unsigned)((ga >> 32) & 0x01ffffffull) | (2u << 30); // addr hi | type=2
  i32x8 g1;
  g1[0] = 1 << 16;                                             // data_size=1 (2 bytes)
  g1[1] = (int)((dim0 & 0xffffu) << 16);                       // tensor_dim0 lo16 @[63:48]
  g1[2] = (int)((dim0 >> 16) | ((dim1 & 0xffffu) << 16));      // dim0 hi16 | tensor_dim1 lo16
  g1[3] = (int)((dim1 >> 16) | ((dim0 & 0xffffu) << 16));      // dim1 hi16 | tile_dim0 @[127:112]
  g1[4] = (int)(dim1 & 0xffffu);                               // tile_dim1 @[143:128]
  g1[5] = (int)stride0;                                        // tensor_dim0_stride lo32
  g1[6] = 0;                                                   // stride0 hi16 | stride1 lo16
  g1[7] = 0;
  i32x4 z4 = {0, 0, 0, 0};
#if __clang_major__ >= 23
  i32x8 z8 = {0, 0, 0, 0, 0, 0, 0, 0};
  __builtin_amdgcn_tensor_load_to_lds(g0, g1, z4, z4, z8, 0);
#else
  __builtin_amdgcn_tensor_load_to_lds(g0, g1, z4, z4, 0);
#endif
}
#endif

// ---------------------------------------------------------------------------
// Kernel 0: transpose + f32->f16 convert the five 256x256 weight matrices.
// WT[n*256+k] = (f16) W[k*256+n]
// ---------------------------------------------------------------------------
__global__ __launch_bounds__(256) void cvt_weights(
    const float* __restrict__ Wq, const float* __restrict__ Wk,
    const float* __restrict__ Wv, const float* __restrict__ Wg,
    const float* __restrict__ Wo,
    half_t* __restrict__ WqT, half_t* __restrict__ WkT,
    half_t* __restrict__ WvT, half_t* __restrict__ WgT,
    half_t* __restrict__ WoT) {
  int idx = blockIdx.x * 256 + threadIdx.x;      // 0..65535
  int n = idx >> 8, k = idx & 255;
  int src = k * 256 + n;
  WqT[idx] = (half_t)Wq[src];
  WkT[idx] = (half_t)Wk[src];
  WvT[idx] = (half_t)Wv[src];
  WgT[idx] = (half_t)Wg[src];
  WoT[idx] = (half_t)Wo[src];
}

// ---------------------------------------------------------------------------
// Kernel 1: LayerNorm over m rows of 256, write f16. One wave per row.
// ---------------------------------------------------------------------------
__global__ __launch_bounds__(256) void ln_m_kernel(
    const float* __restrict__ m, const float* __restrict__ gam,
    const float* __restrict__ bet, half_t* __restrict__ mn) {
  int wave = threadIdx.x >> 5, lane = threadIdx.x & 31;
  size_t row = (size_t)blockIdx.x * 8 + wave;    // 0..32767
  const float* p = m + row * CM + lane * 8;
  float4 x0 = *(const float4*)p;
  float4 x1 = *(const float4*)(p + 4);
  float xs[8] = {x0.x, x0.y, x0.z, x0.w, x1.x, x1.y, x1.z, x1.w};
  float sum = 0.f;
  #pragma unroll
  for (int e = 0; e < 8; ++e) sum += xs[e];
  #pragma unroll
  for (int mask = 1; mask < 32; mask <<= 1) sum += __shfl_xor(sum, mask, 32);
  float mu = sum * (1.f / 256.f);
  float var = 0.f;
  #pragma unroll
  for (int e = 0; e < 8; ++e) { float d = xs[e] - mu; var += d * d; }
  #pragma unroll
  for (int mask = 1; mask < 32; mask <<= 1) var += __shfl_xor(var, mask, 32);
  float rstd = rsqrtf(var * (1.f / 256.f) + 1e-5f);
  union { half_t h[8]; uint4 q; } o;
  #pragma unroll
  for (int e = 0; e < 8; ++e) {
    int c = lane * 8 + e;
    o.h[e] = (half_t)((xs[e] - mu) * rstd * gam[c] + bet[c]);
  }
  *(uint4*)(mn + row * CM + lane * 8) = o.q;
}

// ---------------------------------------------------------------------------
// Kernel 2: LayerNorm over z rows of 128 fused with pair-bias projection.
// One wave per (i,j); output bias[H][R][R] f32.
// ---------------------------------------------------------------------------
__global__ __launch_bounds__(256) void ln_z_bias_kernel(
    const float* __restrict__ z, const float* __restrict__ gam,
    const float* __restrict__ bet, const float* __restrict__ Wz,
    float* __restrict__ bias) {
  __shared__ float wz[CZ * NH];
  for (int t = threadIdx.x; t < CZ * NH; t += 256) wz[t] = Wz[t];
  __syncthreads();
  int wave = threadIdx.x >> 5, lane = threadIdx.x & 31;
  size_t p = (size_t)blockIdx.x * 8 + wave;      // 0..65535 = i*256+j
  float4 x = *(const float4*)(z + p * CZ + lane * 4);
  float xs[4] = {x.x, x.y, x.z, x.w};
  float sum = xs[0] + xs[1] + xs[2] + xs[3];
  #pragma unroll
  for (int mask = 1; mask < 32; mask <<= 1) sum += __shfl_xor(sum, mask, 32);
  float mu = sum * (1.f / 128.f);
  float var = 0.f;
  #pragma unroll
  for (int t = 0; t < 4; ++t) { float d = xs[t] - mu; var += d * d; }
  #pragma unroll
  for (int mask = 1; mask < 32; mask <<= 1) var += __shfl_xor(var, mask, 32);
  float rstd = rsqrtf(var * (1.f / 128.f) + 1e-5f);
  float zn[4];
  #pragma unroll
  for (int t = 0; t < 4; ++t) {
    int c = lane * 4 + t;
    zn[t] = (xs[t] - mu) * rstd * gam[c] + bet[c];
  }
  #pragma unroll
  for (int h = 0; h < NH; ++h) {
    float acc = 0.f;
    #pragma unroll
    for (int t = 0; t < 4; ++t) acc += zn[t] * wz[(lane * 4 + t) * NH + h];
    #pragma unroll
    for (int mask = 1; mask < 32; mask <<= 1) acc += __shfl_xor(acc, mask, 32);
    if (lane == 0) bias[(size_t)h * 65536 + p] = acc;
  }
}

// ---------------------------------------------------------------------------
// Kernel 3: generic WMMA GEMM  C[32768x256] = A[32768x256] * B(=BT N-major).
// Block = 128 rows x 64 cols, 8 waves; wave = 16 rows x 64 cols.
// The 64x256 f16 B tile (32 KB) is DMA'd into LDS once per block by the
// Tensor Data Mover (TENSORcnt), then consumed as ds_load_b128 fragments.
// mode: 0/1 = store Q/K [S,H,R,C] f16, 2 = store V^T [S,H,C,R] f16,
//       3 = sigmoid(x+bg) gate f16 row-major, 4 = f32 out + bo (final GEMM).
// ---------------------------------------------------------------------------
__global__ __launch_bounds__(256) void gemm_wmma(
    const half_t* __restrict__ A, const half_t* __restrict__ BT,
    const float* __restrict__ bias, void* __restrict__ outp, int mode) {
  __shared__ __align__(16) half_t btl[64 * 256];   // 32 KB B tile
  int wave = threadIdx.x >> 5, lane = threadIdx.x & 31;
  int r0 = blockIdx.x * 128 + wave * 16;
  int c0 = blockIdx.y * 64;

#if defined(HAVE_TDM)
  if (threadIdx.x < 32) {
    tdm_stage(BT + (size_t)c0 * 256, btl, 256, 64, 256);
    wait_tensorcnt0();
  }
  __syncthreads();
#else
  {
    const uint4* src = (const uint4*)(BT + (size_t)c0 * 256);
    uint4* dst = (uint4*)btl;
    for (int t = threadIdx.x; t < 64 * 256 / 8; t += 256) dst[t] = src[t];
    __syncthreads();
  }
#endif

  v8f acc[4] = {v8f{}, v8f{}, v8f{}, v8f{}};
  for (int kb = 0; kb < 8; ++kb) {
    v16h a = load_frag16(A, r0, 256, kb * 32);
    if (kb < 7)  // prefetch next A k-chunk into WGP-level caches
      __builtin_prefetch(A + (size_t)r0 * 256 + (kb + 1) * 32, 0, 3);
    #pragma unroll
    for (int j = 0; j < 4; ++j) {
      v16h b = load_frag16(btl, j * 16, 256, kb * 32);   // ds_load_b128 x2
      acc[j] = wmma_f16(a, b, acc[j]);
    }
  }
  int m_lane = 8 * (lane >> 4), n_lane = lane & 15;
  #pragma unroll
  for (int j = 0; j < 4; ++j) {
    #pragma unroll
    for (int e = 0; e < 8; ++e) {
      int r = r0 + e + m_lane;
      int col = c0 + j * 16 + n_lane;
      float v = acc[j][e];
      int s = r >> 8, i = r & 255;
      int h = col >> 5, c = col & 31;
      if (mode <= 1) {
        ((half_t*)outp)[((size_t)(s * 8 + h) * 256 + i) * 32 + c] = (half_t)v;
      } else if (mode == 2) {
        ((half_t*)outp)[((size_t)(s * 8 + h) * 32 + c) * 256 + i] = (half_t)v;
      } else if (mode == 3) {
        v += bias[col];
        v = 1.f / (1.f + __expf(-v));
        ((half_t*)outp)[(size_t)r * 256 + col] = (half_t)v;
      } else {
        ((float*)outp)[(size_t)r * 256 + col] = v + bias[col];
      }
    }
  }
}

// ---------------------------------------------------------------------------
// Kernel 4: flash attention per (s,h). Block = 256 thr = 8 waves, each wave
// owns 32 query rows (two 16-row WMMA tiles). C=32 -> one WMMA per 16x16
// score tile. K [256x32] and V^T [32x256] (16 KB each, contiguous) are
// TDM-staged into LDS once per block, removing the 8x per-wave redundant
// global fragment traffic. Online softmax; P transposed C-layout -> A-layout
// via a 1 KB/wave LDS bounce. Epilogue applies the sigmoid gate and stores
// go = g * o as f16 [S,R,H*C].
// ---------------------------------------------------------------------------
__global__ __launch_bounds__(256) void attn_kernel(
    const half_t* __restrict__ q, const half_t* __restrict__ k,
    const half_t* __restrict__ vt, const float* __restrict__ bias,
    const half_t* __restrict__ gate, half_t* __restrict__ go) {
  __shared__ __align__(16) half_t lds_k[R_RES * CH];   // 16 KB
  __shared__ __align__(16) half_t lds_v[CH * R_RES];   // 16 KB
  __shared__ __align__(16) half_t lds_p[8][16 * 32];   // 8 KB
  int sh = blockIdx.x;                 // s*8 + h
  int s = sh >> 3, h = sh & 7;
  int wave = threadIdx.x >> 5, lane = threadIdx.x & 31;
  const half_t* qb = q + (size_t)sh * R_RES * CH;
  const half_t* kg = k + (size_t)sh * R_RES * CH;
  const half_t* vg = vt + (size_t)sh * CH * R_RES;
  const float*  bb = bias + (size_t)h * R_RES * R_RES;
  half_t* myp = &lds_p[wave][0];
  const float scale = 0.17677669529663687f;   // 1/sqrt(32)
  int m_lane = 8 * (lane >> 4), n_lane = lane & 15;

#if defined(HAVE_TDM)
  if (threadIdx.x < 32) {
    tdm_stage(kg, lds_k, R_RES * CH, 1, R_RES * CH);   // one contiguous row
    tdm_stage(vg, lds_v, CH * R_RES, 1, CH * R_RES);
    wait_tensorcnt0();
  }
  __syncthreads();
#else
  {
    const uint4* ks = (const uint4*)kg;
    const uint4* vs = (const uint4*)vg;
    uint4* kd = (uint4*)lds_k;
    uint4* vd = (uint4*)lds_v;
    for (int t = threadIdx.x; t < R_RES * CH / 8; t += 256) {
      kd[t] = ks[t];
      vd[t] = vs[t];
    }
    __syncthreads();
  }
#endif

  for (int tile = 0; tile < 2; ++tile) {
    int i0 = wave * 32 + tile * 16;
    v16h qa = load_frag16(qb, i0, CH, 0);
    v8f o0 = v8f{}, o1 = v8f{};
    float rmax[8], rsum[8];
    #pragma unroll
    for (int e = 0; e < 8; ++e) { rmax[e] = -1e30f; rsum[e] = 0.f; }

    for (int t = 0; t < 8; ++t) {
      v8f z = v8f{};
      v16h kf0 = load_frag16(lds_k, (2 * t) * 16, CH, 0);
      v8f c0 = wmma_f16(qa, kf0, z);
      v16h kf1 = load_frag16(lds_k, (2 * t + 1) * 16, CH, 0);
      v8f c1 = wmma_f16(qa, kf1, z);

      float nm[8];
      #pragma unroll
      for (int e = 0; e < 8; ++e) {
        int mrow = i0 + e + m_lane;
        c0[e] = c0[e] * scale + bb[(size_t)mrow * R_RES + (2 * t) * 16 + n_lane];
        c1[e] = c1[e] * scale + bb[(size_t)mrow * R_RES + (2 * t + 1) * 16 + n_lane];
        nm[e] = fmaxf(c0[e], c1[e]);
      }
      #pragma unroll
      for (int mask = 1; mask < 16; mask <<= 1) {
        #pragma unroll
        for (int e = 0; e < 8; ++e)
          nm[e] = fmaxf(nm[e], __shfl_xor(nm[e], mask, 32));
      }
      float corr[8], bsum[8];
      #pragma unroll
      for (int e = 0; e < 8; ++e) {
        nm[e] = fmaxf(nm[e], rmax[e]);
        corr[e] = __expf(rmax[e] - nm[e]);
        rmax[e] = nm[e];
        c0[e] = __expf(c0[e] - nm[e]);
        c1[e] = __expf(c1[e] - nm[e]);
        bsum[e] = c0[e] + c1[e];
      }
      #pragma unroll
      for (int mask = 1; mask < 16; mask <<= 1) {
        #pragma unroll
        for (int e = 0; e < 8; ++e)
          bsum[e] += __shfl_xor(bsum[e], mask, 32);
      }
      #pragma unroll
      for (int e = 0; e < 8; ++e) {
        rsum[e] = rsum[e] * corr[e] + bsum[e];
        o0[e] *= corr[e];
        o1[e] *= corr[e];
      }
      // C-layout P -> LDS [16 x 32] row-major, reload as A fragment
      #pragma unroll
      for (int e = 0; e < 8; ++e) {
        myp[(e + m_lane) * 32 + n_lane]      = (half_t)c0[e];
        myp[(e + m_lane) * 32 + 16 + n_lane] = (half_t)c1[e];
      }
      asm volatile("s_wait_dscnt 0" ::: "memory");
      v16h pa = load_frag16(myp, 0, 32, 0);
      v16h v0 = load_frag16(lds_v, 0, R_RES, 32 * t);    // cols c=0..15
      v16h v1 = load_frag16(lds_v, 16, R_RES, 32 * t);   // cols c=16..31
      o0 = wmma_f16(pa, v0, o0);
      o1 = wmma_f16(pa, v1, o1);
    }

    // finalize: divide by row sums, gate, store f16
    const half_t* gb = gate + (size_t)s * R_RES * 256 + h * 32;
    half_t* ob = go + (size_t)s * R_RES * 256 + h * 32;
    #pragma unroll
    for (int e = 0; e < 8; ++e) {
      float inv = 1.f / rsum[e];
      int i = i0 + e + m_lane;
      float g0 = (float)gb[(size_t)i * 256 + n_lane];
      float g1 = (float)gb[(size_t)i * 256 + 16 + n_lane];
      ob[(size_t)i * 256 + n_lane]      = (half_t)(g0 * o0[e] * inv);
      ob[(size_t)i * 256 + 16 + n_lane] = (half_t)(g1 * o1[e] * inv);
    }
  }
}

// ---------------------------------------------------------------------------
// Host launcher
// ---------------------------------------------------------------------------
extern "C" void kernel_launch(void* const* d_in, const int* in_sizes, int n_in,
                              void* d_out, int out_size, void* d_ws, size_t ws_size,
                              hipStream_t stream) {
  const float* m      = (const float*)d_in[0];
  const float* z      = (const float*)d_in[1];
  const float* ln_m_g = (const float*)d_in[2];
  const float* ln_m_b = (const float*)d_in[3];
  const float* ln_z_g = (const float*)d_in[4];
  const float* ln_z_b = (const float*)d_in[5];
  const float* W_z    = (const float*)d_in[6];
  const float* Wq     = (const float*)d_in[7];
  const float* Wk     = (const float*)d_in[8];
  const float* Wv     = (const float*)d_in[9];
  const float* Wg     = (const float*)d_in[10];
  const float* bg     = (const float*)d_in[11];
  const float* Wo     = (const float*)d_in[12];
  const float* bo     = (const float*)d_in[13];

  char* ws = (char*)d_ws;
  const size_t BIG = (size_t)32768 * 256 * 2;   // 16 MiB f16 planes
  half_t* mn_f16 = (half_t*)(ws + 0 * BIG);
  half_t* qbuf   = (half_t*)(ws + 1 * BIG);     // [S,H,R,C]
  half_t* kbuf   = (half_t*)(ws + 2 * BIG);     // [S,H,R,C]
  half_t* vbuf   = (half_t*)(ws + 3 * BIG);     // [S,H,C,R]
  half_t* gbuf   = (half_t*)(ws + 4 * BIG);     // gate [S,R,HC]
  half_t* gobuf  = (half_t*)(ws + 5 * BIG);     // g*o  [S,R,HC]
  float*  bbuf   = (float*)(ws + 6 * BIG);      // bias [H,R,R] f32 (2 MiB)
  half_t* WqT    = (half_t*)(ws + 6 * BIG + 2097152);
  half_t* WkT    = WqT + 65536;
  half_t* WvT    = WkT + 65536;
  half_t* WgT    = WvT + 65536;
  half_t* WoT    = WgT + 65536;

  cvt_weights<<<256, 256, 0, stream>>>(Wq, Wk, Wv, Wg, Wo, WqT, WkT, WvT, WgT, WoT);
  ln_m_kernel<<<4096, 256, 0, stream>>>(m, ln_m_g, ln_m_b, mn_f16);
  ln_z_bias_kernel<<<8192, 256, 0, stream>>>(z, ln_z_g, ln_z_b, W_z, bbuf);

  dim3 gg(256, 4);
  gemm_wmma<<<gg, 256, 0, stream>>>(mn_f16, WqT, nullptr, qbuf, 0);
  gemm_wmma<<<gg, 256, 0, stream>>>(mn_f16, WkT, nullptr, kbuf, 1);
  gemm_wmma<<<gg, 256, 0, stream>>>(mn_f16, WvT, nullptr, vbuf, 2);
  gemm_wmma<<<gg, 256, 0, stream>>>(mn_f16, WgT, bg, gbuf, 3);

  attn_kernel<<<1024, 256, 0, stream>>>(qbuf, kbuf, vbuf, bbuf, gbuf, gobuf);

  gemm_wmma<<<gg, 256, 0, stream>>>(gobuf, WoT, bo, (float*)d_out, 4);
}
